// AttentionModule_69784628626077
// MI455X (gfx1250) — matrix operand order repaired and checked by reading, not compiled, not run on hardware
//
#include <hip/hip_runtime.h>

// ---------------- types ----------------
typedef __attribute__((ext_vector_type(16))) __bf16 v16bf;
typedef __attribute__((ext_vector_type(8)))  __bf16 v8bf;
typedef __attribute__((ext_vector_type(8)))  float  v8f;

#define DEV static __device__ __forceinline__

DEV v16bf concat8(v8bf lo, v8bf hi) {
    return __builtin_shufflevector(lo, hi, 0,1,2,3,4,5,6,7,8,9,10,11,12,13,14,15);
}

// A-matrix fragment (16x32 bf16, MxK): lane row = M; lane holds
// K = hi*16 + [0..15] contiguous  (hi = lane>=16)
DEV v16bf load_a_frag(const __bf16* rowk0, int hi) {
    const v8bf lo = *reinterpret_cast<const v8bf*>(rowk0 + hi * 16);
    const v8bf up = *reinterpret_cast<const v8bf*>(rowk0 + hi * 16 + 8);
    return concat8(lo, up);
}

// B-matrix fragment (32x16 bf16, KxN): lane col = N; lane holds K = hi*16 + [0..15] contiguous
DEV v16bf load_b_frag(const __bf16* colk0, int hi) {
    return *reinterpret_cast<const v16bf*>(colk0 + hi * 16);
}

DEV v8f wmma_bf16(v16bf a, v16bf b, v8f c) {
    return __builtin_amdgcn_wmma_f32_16x16x32_bf16(
        false, a, false, b, (short)0, c, false, false);
}

// ---------------- constants ----------------
constexpr int B_     = 4;
constexpr int T_     = 1024;
constexpr int L_     = 1024;
constexpr int DMODEL = 1024;
constexpr int H_     = 16;
constexpr int HD     = 64;     // head dim
constexpr float INV_SCALE = 0.03125f;              // 1/sqrt(1024), folded into Q
constexpr float LOG1E4_O32 = 0.28782313662425572f; // ln(10000)/32

// ---------------- prep kernels ----------------
__global__ void lens_kernel(const int* __restrict__ xm, const int* __restrict__ cm,
                            float* __restrict__ lens) {
    int which = blockIdx.x;        // 0 = query lens, 1 = key lens
    int b     = blockIdx.y;
    const int* m = which ? cm : xm;
    int s = 0;
    for (int i = threadIdx.x; i < 1024; i += 32) s += m[b * 1024 + i];
    #pragma unroll
    for (int off = 16; off; off >>= 1) s += __shfl_xor(s, off, 32);
    if (threadIdx.x == 0) lens[which * 4 + b] = (float)s;
}

// src: (batch, R, C) f32  ->  dst: (batch, C, R) bf16
__global__ void transpose_convert_kernel(const float* __restrict__ src,
                                         __bf16* __restrict__ dst, int R, int C) {
    __shared__ float tile[32][33];
    size_t boff = (size_t)blockIdx.z * R * C;
    src += boff; dst += boff;
    int c0 = blockIdx.x * 32, r0 = blockIdx.y * 32;
    int tx = threadIdx.x, ty = threadIdx.y;
    #pragma unroll
    for (int i = 0; i < 32; i += 8)
        tile[ty + i][tx] = src[(size_t)(r0 + ty + i) * C + c0 + tx];
    __syncthreads();
    #pragma unroll
    for (int i = 0; i < 32; i += 8)
        dst[(size_t)(c0 + ty + i) * R + r0 + tx] = (__bf16)tile[tx][ty + i];
}

__global__ void convert_kernel(const float* __restrict__ src, __bf16* __restrict__ dst, int n) {
    int i = blockIdx.x * 256 + threadIdx.x;
    if (i < n) dst[i] = (__bf16)src[i];
}

// ---------------- fused projection GEMM (WMMA) ----------------
// Each wave computes a 32x64 tile: 2 row-tiles x 4 col-tiles, 8 WMMAs per k-step.
// MODE 0: +RoPE (query lens) * 1/32, write bf16 (B,T,1024)
// MODE 1: +RoPE (key lens),         write bf16 (B,L,1024)
// MODE 2: write V transposed bf16 (B,H,64,L)
// MODE 3: output projection: +bias, query-mask, store f32 transposed (B,DMODEL,T)
template <int MODE>
__global__ void gemm_wmma_kernel(const __bf16* __restrict__ A,
                                 const __bf16* __restrict__ Wt,
                                 const float*  __restrict__ bias,
                                 void* __restrict__ out,
                                 const float* __restrict__ lens) {
    const int wave = threadIdx.x >> 5;
    const int lane = threadIdx.x & 31;
    const int ln   = lane & 15;
    const int hi   = lane >> 4;
    const int mBase = (blockIdx.x * 8 + wave) * 32;   // 32 rows per wave
    const int nBase = blockIdx.y * 64;

    v8f zero = {};
    v8f acc[2][4] = {{zero, zero, zero, zero}, {zero, zero, zero, zero}};

    for (int k0 = 0; k0 < 1024; k0 += 32) {
        v16bf af0 = load_a_frag(A + (size_t)(mBase + ln) * 1024 + k0, hi);
        v16bf af1 = load_a_frag(A + (size_t)(mBase + 16 + ln) * 1024 + k0, hi);
        #pragma unroll
        for (int j = 0; j < 4; j++) {
            v16bf bf = load_b_frag(Wt + (size_t)(nBase + j * 16 + ln) * 1024 + k0, hi);
            acc[0][j] = wmma_bf16(af0, bf, acc[0][j]);
            acc[1][j] = wmma_bf16(af1, bf, acc[1][j]);
        }
    }

    #pragma unroll
    for (int mt = 0; mt < 2; mt++) {
        const int mB = mBase + mt * 16;
        if (MODE == 0 || MODE == 1) {
            __bf16* outp = (__bf16*)out;
            const float post = (MODE == 0) ? INV_SCALE : 1.0f;
            #pragma unroll
            for (int jp = 0; jp < 2; jp++) {
                int n1 = nBase + jp * 16 + ln;   // first half of head pair
                int n2 = n1 + 32;                // partner
                float b1 = bias[n1], b2 = bias[n2];
                int i = jp * 16 + ln;            // rotary index 0..31
                float th = 10.0f * __expf(-LOG1E4_O32 * (float)i);
                #pragma unroll
                for (int v = 0; v < 8; v++) {
                    int r = mB + v + hi * 8;
                    int bidx = r >> 10, pos = r & 1023;
                    float fr = ((float)pos / lens[bidx]) * th;
                    float c = __cosf(fr), s = __sinf(fr);
                    float x1 = acc[mt][jp][v] + b1;
                    float x2 = acc[mt][jp + 2][v] + b2;
                    outp[(size_t)r * 1024 + n1] = (__bf16)((x1 * c - x2 * s) * post);
                    outp[(size_t)r * 1024 + n2] = (__bf16)((x1 * s + x2 * c) * post);
                }
            }
        } else if (MODE == 2) {
            __bf16* outp = (__bf16*)out;
            #pragma unroll
            for (int j = 0; j < 4; j++) {
                int dg = nBase + j * 16 + ln;
                int h = dg >> 6, d = dg & 63;
                float bb = bias[dg];
                #pragma unroll
                for (int v = 0; v < 8; v++) {
                    int r = mB + v + hi * 8;
                    int bidx = r >> 10, l = r & 1023;
                    outp[((size_t)((bidx * H_ + h) * HD + d)) * L_ + l] =
                        (__bf16)(acc[mt][j][v] + bb);
                }
            }
        } else {
            float* outp = (float*)out;
            #pragma unroll
            for (int j = 0; j < 4; j++) {
                int n = nBase + j * 16 + ln;
                float bb = bias[n];
                #pragma unroll
                for (int v = 0; v < 8; v++) {
                    int r = mB + v + hi * 8;
                    int bidx = r >> 10, t = r & 1023;
                    float keep = ((float)t < lens[bidx]) ? 1.0f : 0.0f;
                    outp[((size_t)(bidx * DMODEL + n)) * T_ + t] =
                        (acc[mt][j][v] + bb) * keep;
                }
            }
        }
    }
}

// ---------------- flash attention (WMMA, online softmax) ----------------
// grid (T/64, B*H), block 128: each wave handles 16 queries of one (b,h).
// Q is pre-scaled by 1/sqrt(d). Key loop stops at ceil(len_k/32); only the
// tail chunk applies the padding mask (uniform branch, EXEC stays full).
__global__ void attention_kernel(const __bf16* __restrict__ qbf,
                                 const __bf16* __restrict__ kbf,
                                 const __bf16* __restrict__ vT,
                                 __bf16* __restrict__ attn_out,
                                 const float* __restrict__ lens_k) {
    __shared__ __align__(16) __bf16 pbuf[4][16 * 32];
    const int wave = threadIdx.x >> 5;
    const int lane = threadIdx.x & 31;
    const int ln   = lane & 15;
    const int hi   = lane >> 4;
    const int bh = blockIdx.y;
    const int b = bh >> 4, h = bh & 15;
    const int qBase = blockIdx.x * 64 + wave * 16;
    const int lenk  = (int)lens_k[b];
    const int lkEnd = (lenk + 31) & ~31;   // uniform per block

    // Q^T used as B-matrix: column = query (contiguous along head dim in memory)
    v16bf qf[2];
    #pragma unroll
    for (int s = 0; s < 2; s++)
        qf[s] = load_b_frag(qbf + (size_t)(b * T_ + qBase + ln) * 1024 + h * HD + s * 32, hi);

    v8f zero = {};
    v8f o[4] = {zero, zero, zero, zero};
    float m_run = -__builtin_inff();
    float l_run = 0.0f;

    for (int l0 = 0; l0 < lkEnd; l0 += 32) {
        // S^T tiles: rows = keys (2 tiles of 16), cols = 16 queries.
        // t2 inner so the two accumulator chains alternate (hazard distance 2).
        v8f st[2] = {zero, zero};
        #pragma unroll
        for (int s = 0; s < 2; s++) {
            #pragma unroll
            for (int t2 = 0; t2 < 2; t2++) {
                v16bf ka = load_a_frag(
                    kbf + (size_t)(b * L_ + l0 + t2 * 16 + ln) * 1024 + h * HD + s * 32, hi);
                st[t2] = wmma_bf16(ka, qf[s], st[t2]);
            }
        }
        float sv[16];
        #pragma unroll
        for (int t2 = 0; t2 < 2; t2++)
            #pragma unroll
            for (int v = 0; v < 8; v++) sv[t2 * 8 + v] = st[t2][v];
        if (l0 + 32 > lenk) {   // uniform branch: only the tail chunk masks
            #pragma unroll
            for (int t2 = 0; t2 < 2; t2++)
                #pragma unroll
                for (int v = 0; v < 8; v++) {
                    int key = l0 + t2 * 16 + hi * 8 + v;
                    if (key >= lenk) sv[t2 * 8 + v] = -__builtin_inff();
                }
        }
        // online softmax over key axis (per-lane + one cross-half shuffle)
        float cm = -__builtin_inff();
        #pragma unroll
        for (int i = 0; i < 16; i++) cm = fmaxf(cm, sv[i]);
        cm = fmaxf(cm, __shfl_xor(cm, 16, 32));
        float m_new = fmaxf(m_run, cm);
        float scale = __expf(m_run - m_new);
        float p[16], psum = 0.0f;
        #pragma unroll
        for (int i = 0; i < 16; i++) { p[i] = __expf(sv[i] - m_new); psum += p[i]; }
        psum += __shfl_xor(psum, 16, 32);
        l_run = l_run * scale + psum;
        m_run = m_new;
        // rescale accumulator rows (row = query = vgpr + 8*hi; stats live in lane (query&15))
        float scl[8];
        #pragma unroll
        for (int v = 0; v < 8; v++) scl[v] = __shfl(scale, v + hi * 8, 32);
        #pragma unroll
        for (int j = 0; j < 4; j++)
            #pragma unroll
            for (int v = 0; v < 8; v++) o[j][v] *= scl[v];
        // P^T (keys x queries) -> LDS row-major P[query][key] for A-fragment reload
        #pragma unroll
        for (int t2 = 0; t2 < 2; t2++) {
            v8bf pk;
            #pragma unroll
            for (int v = 0; v < 8; v++) pk[v] = (__bf16)p[t2 * 8 + v];
            *reinterpret_cast<v8bf*>(&pbuf[wave][ln * 32 + t2 * 16 + hi * 8]) = pk;
        }
        __builtin_amdgcn_wave_barrier();
        v16bf pa = load_a_frag(&pbuf[wave][ln * 32], hi);
        // O += P @ V  (V pre-transposed per head: columns contiguous along keys)
        #pragma unroll
        for (int j = 0; j < 4; j++) {
            v16bf vb = load_b_frag(
                vT + (size_t)((b * H_ + h) * HD + j * 16 + ln) * L_ + l0, hi);
            o[j] = wmma_bf16(pa, vb, o[j]);
        }
        __builtin_amdgcn_wave_barrier();
    }

    float linv[8];
    #pragma unroll
    for (int v = 0; v < 8; v++) linv[v] = 1.0f / __shfl(l_run, v + hi * 8, 32);
    #pragma unroll
    for (int j = 0; j < 4; j++)
        #pragma unroll
        for (int v = 0; v < 8; v++) {
            int r = qBase + v + hi * 8;
            attn_out[(size_t)(b * T_ + r) * 1024 + h * HD + j * 16 + ln] =
                (__bf16)(o[j][v] * linv[v]);
        }
}

// ---------------- host launch ----------------
extern "C" void kernel_launch(void* const* d_in, const int* in_sizes, int n_in,
                              void* d_out, int out_size, void* d_ws, size_t ws_size,
                              hipStream_t stream) {
    (void)in_sizes; (void)n_in; (void)out_size; (void)ws_size;
    const float* x       = (const float*)d_in[0];
    const float* context = (const float*)d_in[1];
    const int*   x_mask  = (const int*)d_in[2];
    const int*   c_mask  = (const int*)d_in[3];
    const float* Wq = (const float*)d_in[4];  const float* bq = (const float*)d_in[5];
    const float* Wk = (const float*)d_in[6];  const float* bk = (const float*)d_in[7];
    const float* Wv = (const float*)d_in[8];  const float* bv = (const float*)d_in[9];
    const float* Wo = (const float*)d_in[10]; const float* bo = (const float*)d_in[11];
    float* out = (float*)d_out;

    constexpr size_t MB = 1ull << 20;
    char* ws = (char*)d_ws;
    __bf16* xT   = (__bf16*)(ws + 0 * MB);   // 8 MiB  (B,T,DMODEL) bf16
    __bf16* ctxb = (__bf16*)(ws + 8 * MB);   // 8 MiB  (B,L,DCTX) bf16
    __bf16* Wqt  = (__bf16*)(ws + 16 * MB);  // 2 MiB each, transposed bf16
    __bf16* Wkt  = (__bf16*)(ws + 18 * MB);
    __bf16* Wvt  = (__bf16*)(ws + 20 * MB);
    __bf16* Wot  = (__bf16*)(ws + 22 * MB);
    __bf16* qbf  = (__bf16*)(ws + 24 * MB);  // 8 MiB roped+scaled Q bf16
    __bf16* kbf  = (__bf16*)(ws + 32 * MB);  // 8 MiB roped K bf16
    __bf16* vTb  = (__bf16*)(ws + 40 * MB);  // 8 MiB V transposed (B,H,64,L)
    __bf16* aout = (__bf16*)(ws + 48 * MB);  // 8 MiB attention output bf16
    float*  lens = (float*)(ws + 56 * MB);   // [0..3]=len_q, [4..7]=len_k

    lens_kernel<<<dim3(2, 4), 32, 0, stream>>>(x_mask, c_mask, lens);

    // x (B, DMODEL, T) -> xT (B, T, DMODEL) bf16
    transpose_convert_kernel<<<dim3(32, 32, 4), dim3(32, 8), 0, stream>>>(x, xT, 1024, 1024);
    // W (K,N) -> Wt (N,K) bf16
    transpose_convert_kernel<<<dim3(32, 32, 1), dim3(32, 8), 0, stream>>>(Wq, Wqt, 1024, 1024);
    transpose_convert_kernel<<<dim3(32, 32, 1), dim3(32, 8), 0, stream>>>(Wk, Wkt, 1024, 1024);
    transpose_convert_kernel<<<dim3(32, 32, 1), dim3(32, 8), 0, stream>>>(Wv, Wvt, 1024, 1024);
    transpose_convert_kernel<<<dim3(32, 32, 1), dim3(32, 8), 0, stream>>>(Wo, Wot, 1024, 1024);
    convert_kernel<<<(B_ * L_ * 1024) / 256, 256, 0, stream>>>(context, ctxb, B_ * L_ * 1024);

    // Projections (fused bias + RoPE / transpose epilogues); 32x64 tile per wave
    gemm_wmma_kernel<0><<<dim3(16, 16), 256, 0, stream>>>(xT,   Wqt, bq, (void*)qbf, lens);
    gemm_wmma_kernel<1><<<dim3(16, 16), 256, 0, stream>>>(ctxb, Wkt, bk, (void*)kbf, lens + 4);
    gemm_wmma_kernel<2><<<dim3(16, 16), 256, 0, stream>>>(ctxb, Wvt, bv, (void*)vTb, lens);

    // Attention
    attention_kernel<<<dim3(16, 64), 128, 0, stream>>>(qbf, kbf, vTb, aout, lens + 4);

    // Output projection + query mask + transposed store
    gemm_wmma_kernel<3><<<dim3(16, 16), 256, 0, stream>>>(aout, Wot, bo, (void*)out, lens);
}
